// ResonanceLayer_19370302505439
// MI455X (gfx1250) — matrix operand order repaired and checked by reading, not compiled, not run on hardware
//
#include <hip/hip_runtime.h>
#include <hip/hip_bf16.h>

typedef __attribute__((ext_vector_type(16))) _Float16 v16h;
typedef __attribute__((ext_vector_type(8)))  _Float16 v8h;
typedef __attribute__((ext_vector_type(4)))  _Float16 v4h;
typedef __attribute__((ext_vector_type(8)))  float    v8f;
typedef __attribute__((ext_vector_type(4)))  float    v4f;

#define NB   1024
#define NN   64
#define NT   8
#define ND   128
#define NDH  128
#define NDO  64
#define ROWS_PER_WG 128
#define XSTR 136                    // padded half-stride for LDS activations
#define TWO_PI_F 6.283185307179586f

// LDS region sizes in halves
#define W1_H (32*512)               // 8 col-tiles x 4 k-chunks x 32 lanes x 16 halves
#define W2_H (32*512)
#define W3_H (16*512)               // 4 col-tiles x 4 k-chunks
#define ACT_H (ROWS_PER_WG*XSTR)

#define SMEM_BYTES (((W1_H)+(W2_H)+(W3_H)+2*(ACT_H))*2 + 320*4)

// ---- A fragment: 16x32 f16 tile, contiguous-k layout (matches B packing): lane holds row M=lane%16,
//      halves 0..15 = K(g*16 .. g*16+15)
static __device__ inline v16h lds_load_a(const _Float16* act, int rowBase, int lane, int kt) {
    const int row = rowBase + (lane & 15);
    const int g   = lane >> 4;
    const _Float16* p = act + row * XSTR + kt * 32 + g * 16;
    v8h lo = *(const v8h*)(p);
    v8h hi = *(const v8h*)(p + 8);
    v16h a;
#pragma unroll
    for (int i = 0; i < 8; ++i) { a[i] = lo[i]; a[i + 8] = hi[i]; }
    return a;
}

// ---- B fragment: pre-swizzled in LDS; 32 contiguous bytes per lane
static __device__ inline v16h lds_load_b(const _Float16* wbuf, int chunk, int lane) {
    const _Float16* p = wbuf + chunk * 512 + lane * 16;
    v8h lo = *(const v8h*)(p);
    v8h hi = *(const v8h*)(p + 8);
    v16h b;
#pragma unroll
    for (int i = 0; i < 8; ++i) { b[i] = lo[i]; b[i + 8] = hi[i]; }
    return b;
}

template<int NCT, bool TO_GLOBAL>
static __device__ inline void mlp_layer(const _Float16* actIn, const _Float16* wbuf,
                                        const float* bias, _Float16* actOut,
                                        float* gOut, long gRow0,
                                        int rowBase, int lane) {
    v16h a0 = lds_load_a(actIn, rowBase, lane, 0);
    v16h a1 = lds_load_a(actIn, rowBase, lane, 1);
    v16h a2 = lds_load_a(actIn, rowBase, lane, 2);
    v16h a3 = lds_load_a(actIn, rowBase, lane, 3);
    const int g = lane >> 4;
#pragma unroll
    for (int ct = 0; ct < NCT; ++ct) {
        v8f acc = {};
        acc = __builtin_amdgcn_wmma_f32_16x16x32_f16(false, a0, false,
                lds_load_b(wbuf, ct * 4 + 0, lane), (short)0, acc, false, false);
        acc = __builtin_amdgcn_wmma_f32_16x16x32_f16(false, a1, false,
                lds_load_b(wbuf, ct * 4 + 1, lane), (short)0, acc, false, false);
        acc = __builtin_amdgcn_wmma_f32_16x16x32_f16(false, a2, false,
                lds_load_b(wbuf, ct * 4 + 2, lane), (short)0, acc, false, false);
        acc = __builtin_amdgcn_wmma_f32_16x16x32_f16(false, a3, false,
                lds_load_b(wbuf, ct * 4 + 3, lane), (short)0, acc, false, false);
        const int nc = ct * 16 + (lane & 15);
        const float bb = bias[nc];
#pragma unroll
        for (int i = 0; i < 8; ++i) {
            float vv = acc[i] + bb;
            vv = vv > 0.0f ? vv : 0.0f;
            const int row = rowBase + g * 8 + i;
            if constexpr (TO_GLOBAL) {
                gOut[(gRow0 + row) * (long)NDO + nc] = vv;
            } else {
                actOut[row * XSTR + nc] = (_Float16)vv;
            }
        }
    }
}

__global__ __launch_bounds__(256)
void resonance_mlp_kernel(const float* __restrict__ f_ego, const float* __restrict__ f_nei,
                          const float* __restrict__ W1, const float* __restrict__ b1,
                          const float* __restrict__ W2, const float* __restrict__ b2,
                          const float* __restrict__ W3, const float* __restrict__ b3,
                          float* __restrict__ f_re) {
    extern __shared__ char smem_raw[];
    _Float16* w1s = (_Float16*)smem_raw;
    _Float16* w2s = w1s + W1_H;
    _Float16* w3s = w2s + W2_H;
    _Float16* xbf = w3s + W3_H;
    _Float16* h1f = xbf + ACT_H;
    float* b1s = (float*)(h1f + ACT_H);
    float* b2s = b1s + 128;
    float* b3s = b2s + 128;

    const int tid = threadIdx.x;

    // ---- Stage weights as f16 in B-fragment layout. Iterate in GLOBAL
    //      row-major order (coalesced float4 loads; consecutive lanes read
    //      consecutive columns n) and scatter cheap 16-bit stores into LDS.
    //      Element (k, n) lives at LDS half index
    //      ((ct*4 + kt)*32 + l)*16 + j, with ct=n>>4, kt=k>>5,
    //      l=(n&15)+((k>>4)&1)*16, j=k&15.
#pragma unroll 1
    for (int i4 = tid; i4 < (NDH * NDH) / 4; i4 += 256) {
        const int idx = i4 * 4;
        const int k  = idx >> 7;          // row of W (K dim)
        const int n0 = idx & 127;         // 4-aligned column
        const int kt = k >> 5;
        const int j  = k & 15;
        const int lh = (k >> 4) & 1;
        const int ct = n0 >> 4;           // same for all 4 columns (16 % 4 == 0)
        const int dst0 = ((ct * 4 + kt) * 32 + (n0 & 15) + lh * 16) * 16 + j;
        const v4f w1v = *(const v4f*)(W1 + idx);
        const v4f w2v = *(const v4f*)(W2 + idx);
#pragma unroll
        for (int q = 0; q < 4; ++q) {
            w1s[dst0 + q * 16] = (_Float16)w1v[q];
            w2s[dst0 + q * 16] = (_Float16)w2v[q];
        }
    }
#pragma unroll 1
    for (int i4 = tid; i4 < (NDH * NDO) / 4; i4 += 256) {
        const int idx = i4 * 4;
        const int k  = idx >> 6;
        const int n0 = idx & 63;
        const int kt = k >> 5;
        const int j  = k & 15;
        const int lh = (k >> 4) & 1;
        const int ct = n0 >> 4;
        const int dst0 = ((ct * 4 + kt) * 32 + (n0 & 15) + lh * 16) * 16 + j;
        const v4f w3v = *(const v4f*)(W3 + idx);
#pragma unroll
        for (int q = 0; q < 4; ++q) {
            w3s[dst0 + q * 16] = (_Float16)w3v[q];
        }
    }
    if (tid < 128) { b1s[tid] = b1[tid]; b2s[tid] = b2[tid]; }
    if (tid < 64)  { b3s[tid] = b3[tid]; }

    // ---- Stage activations: f = f_ego (broadcast over N) * f_nei, as f16.
    //      Thread tid covers row tid/2, column-half tid%2 -> each wave stages
    //      exactly its own 16 rows. f_nei/f_ego are streamed non-temporally
    //      so the 256MB pass does not evict the L2-resident f_re intermediate.
    const long r0 = (long)blockIdx.x * ROWS_PER_WG;
    {
        const int  lr = tid >> 1;
        const int  hh = tid & 1;
        const long r  = r0 + lr;
        const int  b  = (int)(r >> 9);              // / (N*T) = 512
        const int  t  = (int)(r & 7);
        const float* np = f_nei + r * (long)ND + hh * 64;
        const float* ep = f_ego + ((long)b * NT + t) * ND + hh * 64;
        _Float16* xr = xbf + lr * XSTR + hh * 64;
#pragma unroll
        for (int c = 0; c < 16; ++c) {
            v4f a = __builtin_nontemporal_load((const v4f*)(np + c * 4));
            v4f e = __builtin_nontemporal_load((const v4f*)(ep + c * 4));
            v4h h;
#pragma unroll
            for (int q = 0; q < 4; ++q) h[q] = (_Float16)(a[q] * e[q]);
            *(v4h*)(xr + c * 4) = h;
        }
    }
    __syncthreads();   // only needed for the shared weight staging

    const int wv = tid >> 5, lane = tid & 31;
    const int rowBase = wv * 16;
    // X -> H1 -> X(reused) -> f_re   (each wave touches only its own rows;
    // intra-wave LDS RAW is hardware-ordered)
    mlp_layer<8, false>(xbf, w1s, b1s, h1f, nullptr, 0,  rowBase, lane);
    mlp_layer<8, false>(h1f, w2s, b2s, xbf, nullptr, 0,  rowBase, lane);
    mlp_layer<4, true >(xbf, w3s, b3s, nullptr, f_re, r0, rowBase, lane);
}

// ---- Kernel 2: angular partition aggregation + tiny 2->64 matmul ----
__global__ __launch_bounds__(256)
void resonance_agg_kernel(const float* __restrict__ x_ego, const float* __restrict__ x_nei,
                          const float* __restrict__ f_re,
                          const float* __restrict__ Wce, const float* __restrict__ bce,
                          float* __restrict__ out_m) {
    const int lane = threadIdx.x & 31;
    const int pi   = blockIdx.x * 8 + (threadIdx.x >> 5);   // (b,t) pair, 0..8191
    const int b = pi >> 3, t = pi & 7;
    const int d0 = lane * 2;

    const float wa0 = Wce[d0],      wa1 = Wce[d0 + 1];
    const float wb0 = Wce[64 + d0], wb1 = Wce[64 + d0 + 1];
    const float bc0 = bce[d0],      bc1 = bce[d0 + 1];

    float acc0[8], acc1[8], cnt[8], dsum[8], asum[8];
#pragma unroll
    for (int p = 0; p < 8; ++p) { acc0[p] = acc1[p] = cnt[p] = dsum[p] = asum[p] = 0.0f; }

    const float ex = x_ego[((long)b * NT + t) * 2 + 0];
    const float ey = x_ego[((long)b * NT + t) * 2 + 1];

    for (int n = 0; n < NN; ++n) {
        const long gi = ((long)b * NN + n) * NT + t;
        const float p0 = x_nei[gi * 2 + 0] - ex;
        const float p1 = x_nei[gi * 2 + 1] - ey;
        const float dist = sqrtf(p0 * p0 + p1 * p1);
        float ang = atan2f(p0, p1);                // arctan2(p[...,0], p[...,1])
        if (ang < 0.0f) ang += TWO_PI_F;           // % 2*pi
        const int  idx = (int)(ang * (8.0f / TWO_PI_F));
        const bool m   = ((fabsf(p0) + fabsf(p1)) > 0.05f) && (dist > 0.005f);
        const float* fr = f_re + gi * (long)NDO + d0;
        const float f0 = fr[0], f1 = fr[1];
#pragma unroll
        for (int p = 0; p < 8; ++p) {
            const float sel = (m && (p == idx)) ? 1.0f : 0.0f;
            cnt[p]  += sel;
            dsum[p] += sel * dist;
            asum[p] += sel * ang;
            acc0[p] += sel * f0;
            acc1[p] += sel * f1;
        }
    }
#pragma unroll
    for (int p = 0; p < 8; ++p) {
        const float dn = 1.0f / (cnt[p] + 1e-4f);
        const long base = (((long)b * NT + t) * 8 + p) * 128;
        out_m[base + d0]     = acc0[p] * dn;
        out_m[base + d0 + 1] = acc1[p] * dn;
        const float dp = dsum[p] * dn, ap = asum[p] * dn;
        float o0 = dp * wa0 + ap * wb0 + bc0;
        float o1 = dp * wa1 + ap * wb1 + bc1;
        out_m[base + 64 + d0]     = o0 > 0.0f ? o0 : 0.0f;
        out_m[base + 64 + d0 + 1] = o1 > 0.0f ? o1 : 0.0f;
    }
}

extern "C" void kernel_launch(void* const* d_in, const int* in_sizes, int n_in,
                              void* d_out, int out_size, void* d_ws, size_t ws_size,
                              hipStream_t stream) {
    const float* x_ego = (const float*)d_in[0];
    const float* x_nei = (const float*)d_in[1];
    const float* f_ego = (const float*)d_in[2];
    const float* f_nei = (const float*)d_in[3];
    const float* W1    = (const float*)d_in[4];
    const float* b1    = (const float*)d_in[5];
    const float* W2    = (const float*)d_in[6];
    const float* b2    = (const float*)d_in[7];
    const float* W3    = (const float*)d_in[8];
    const float* b3    = (const float*)d_in[9];
    const float* Wce   = (const float*)d_in[10];
    const float* bce   = (const float*)d_in[11];

    float* out  = (float*)d_out;
    float* f_re = out + (long)NB * NT * 8 * 128;   // 8,388,608 floats of re_matrix first

    const int total_rows = NB * NN * NT;           // 524288
    const int grid1 = total_rows / ROWS_PER_WG;    // 4096

    resonance_mlp_kernel<<<grid1, 256, SMEM_BYTES, stream>>>(
        f_ego, f_nei, W1, b1, W2, b2, W3, b3, f_re);

    const int grid2 = (NB * NT) / 8;               // 1024 blocks, 8 waves each
    resonance_agg_kernel<<<grid2, 256, 0, stream>>>(
        x_ego, x_nei, f_re, Wce, bce, out);
}
